// WESS_Encoder_60936995995761
// MI455X (gfx1250) — matrix operand order repaired and checked by reading, not compiled, not run on hardware
//
#include <hip/hip_runtime.h>
#include <cstddef>

// ---------------------------------------------------------------------------
// CDNA5 / gfx1250 WESS encoder.
// All dense algebra runs through v_wmma_f32_16x16x32_bf16 (wave32 WMMA).
// GRU recurrences are persistent single-workgroup kernels with h in LDS.
// ---------------------------------------------------------------------------

typedef __attribute__((ext_vector_type(16))) __bf16 v16bf;
typedef __attribute__((ext_vector_type(16))) unsigned short v16u;
typedef __attribute__((ext_vector_type(8)))  float v8f;

namespace {
constexpr int kB = 16, kL = 512, kW = 64, kEMB = 256, kGRUH = 768;
constexpr int kPREH = 1024, kPREO = 256, kBANKC = 128, kBANKK = 8;
}

__device__ __forceinline__ unsigned short f2bf(float f) {
  unsigned int u = __float_as_uint(f);
  u += 0x7FFFu + ((u >> 16) & 1u);   // round-to-nearest-even
  return (unsigned short)(u >> 16);
}
__device__ __forceinline__ float sigmoidf(float x) {
  return 1.f / (1.f + __expf(-x));
}

// ---- elementwise helpers ---------------------------------------------------

__global__ void k_cvt_bf16(const float* __restrict__ s, unsigned short* __restrict__ d, int n) {
  int i = blockIdx.x * blockDim.x + threadIdx.x;
  if (i < n) d[i] = f2bf(s[i]);
}

__global__ void k_zero(float* __restrict__ p, int n) {
  int i = blockIdx.x * blockDim.x + threadIdx.x;
  if (i < n) p[i] = 0.f;
}

// emb gather: ef = table[x], eb = bf16(ef).  grid exactly B*L*EMB threads.
__global__ void k_embed(const int* __restrict__ x, const float* __restrict__ tab,
                        float* __restrict__ ef, unsigned short* __restrict__ eb) {
  int i = blockIdx.x * blockDim.x + threadIdx.x;
  int c = i & (kEMB - 1);
  int bl = i >> 8;
  float v = tab[x[bl] * kEMB + c];
  ef[i] = v;
  eb[i] = f2bf(v);
}

// scatter 1.0 at word starts / ends into (B,L) masks (pre-zeroed)
__global__ void k_masks(const int* __restrict__ gate, float* __restrict__ sm,
                        float* __restrict__ em) {
  int i = blockIdx.x * blockDim.x + threadIdx.x;
  if (i >= kB * kW) return;
  int b = i / kW, w = i % kW;
  sm[b * kL + gate[b * (kW + 1) + w]] = 1.f;
  em[b * kL + gate[b * (kW + 1) + w + 1] - 1] = 1.f;
}

// ---- generic bf16 WMMA GEMM:  C[M,N] = act(A[M,K] @ W[N,K]^T + bias) -------
// A,W row-major bf16 (ushort bits).  One wave per 16x16 C tile, 8 waves/block.
__global__ void k_gemm_bf16(const unsigned short* __restrict__ A, int lda,
                            const unsigned short* __restrict__ Wt, int ldw,
                            const float* __restrict__ bias,
                            float* __restrict__ C, int ldc,
                            int M, int N, int K, int act) {
  const int wave = threadIdx.x >> 5, lane = threadIdx.x & 31;
  const int half = lane >> 4, lm = lane & 15;
  const int ntn = N >> 4;
  long tile = (long)blockIdx.x * (blockDim.x >> 5) + wave;
  long total = (long)(M >> 4) * ntn;
  if (tile >= total) return;                 // wave-uniform exit (EXEC stays full)
  const int m0 = (int)(tile / ntn) << 4;
  const int n0 = (int)(tile % ntn) << 4;
  const unsigned short* ar = A + (size_t)(m0 + lm) * lda;
  const unsigned short* wr = Wt + (size_t)(n0 + lm) * ldw;
  v8f acc = {};
  for (int k0 = 0; k0 < K; k0 += 32) {
    v16u au, bu;
    const int ka = k0 + (half << 4);
#pragma unroll
    for (int e = 0; e < 16; ++e) au[e] = ar[ka + e];
    const int kb = k0 + (half << 4);
#pragma unroll
    for (int e = 0; e < 16; ++e) bu[e] = wr[kb + e];
    acc = __builtin_amdgcn_wmma_f32_16x16x32_bf16(
        false, __builtin_bit_cast(v16bf, au),
        false, __builtin_bit_cast(v16bf, bu),
        (short)0, acc, false, false);
  }
  const int col = n0 + lm;
  const float bv = bias ? bias[col] : 0.f;
#pragma unroll
  for (int i = 0; i < 8; ++i) {
    float v = acc[i] + bv;
    if (act == 1) v = fmaxf(v, 0.f);
    else if (act == 2) v = sigmoidf(v);
    C[(size_t)(m0 + (half << 3) + i) * ldc + col] = v;
  }
}

// ---- persistent GRU scan (one workgroup, 32 waves, h in LDS) ---------------
// Gi: precomputed x@Wi^T+bi, rows (b*T+t), width 3H.  WhT: (3H,H) bf16.
// Each wave owns 16-column groups of H; per step it computes r/z/n gate
// blocks with three fused WMMA accumulators and applies the nonlinearity
// directly on the accumulator registers.  out[b, tt, ocol + j] (+=|=) h.
template <int H>
__global__ void __launch_bounds__(1024) k_gru_scan(
    const float* __restrict__ Gi, const unsigned short* __restrict__ WhT,
    const float* __restrict__ bh, const float* __restrict__ mask,
    int T, int rev, float* __restrict__ out, long obstride, int ostride,
    int ocol, int accum) {
  __shared__ float hst[16 * H];
  __shared__ unsigned short hbf[16 * H];
  const int tid = threadIdx.x, nthr = blockDim.x;
  const int wave = tid >> 5, lane = tid & 31, half = lane >> 4, lm = lane & 15;
  const int nwaves = nthr >> 5;
  const int ngrp = H >> 4;
  for (int i = tid; i < 16 * H; i += nthr) hst[i] = 0.f;
  __syncthreads();
  for (int t = 0; t < T; ++t) {
    const int tt = rev ? (T - 1 - t) : t;
    // phase 1: optional reset h *= (1-r_t), refresh bf16 copy of h
    for (int i = tid; i < 16 * H; i += nthr) {
      float hv = hst[i];
      if (mask) { hv *= (1.f - mask[(i / H) * T + tt]); hst[i] = hv; }
      hbf[i] = f2bf(hv);
    }
    __syncthreads();
    // phase 2: fused  gh = h @ Wh^T  +  gate math  +  h update
    for (int g = wave; g < ngrp; g += nwaves) {
      const int j0 = g << 4;
      const unsigned short* wrR = WhT + (size_t)(j0 + lm) * H;
      const unsigned short* wrZ = WhT + (size_t)(H + j0 + lm) * H;
      const unsigned short* wrN = WhT + (size_t)(2 * H + j0 + lm) * H;
      const unsigned short* arw = &hbf[lm * H];
      v8f aR = {}, aZ = {}, aN = {};
      for (int k0 = 0; k0 < H; k0 += 32) {
        v16u au, bu;
        const int ka = k0 + (half << 4);
#pragma unroll
        for (int e = 0; e < 16; ++e) au[e] = arw[ka + e];
        const v16bf a = __builtin_bit_cast(v16bf, au);
        const int kb = k0 + (half << 4);
#pragma unroll
        for (int e = 0; e < 16; ++e) bu[e] = wrR[kb + e];
        aR = __builtin_amdgcn_wmma_f32_16x16x32_bf16(
            false, a, false, __builtin_bit_cast(v16bf, bu), (short)0, aR, false, false);
#pragma unroll
        for (int e = 0; e < 16; ++e) bu[e] = wrZ[kb + e];
        aZ = __builtin_amdgcn_wmma_f32_16x16x32_bf16(
            false, a, false, __builtin_bit_cast(v16bf, bu), (short)0, aZ, false, false);
#pragma unroll
        for (int e = 0; e < 16; ++e) bu[e] = wrN[kb + e];
        aN = __builtin_amdgcn_wmma_f32_16x16x32_bf16(
            false, a, false, __builtin_bit_cast(v16bf, bu), (short)0, aN, false, false);
      }
      const int j = j0 + lm;
      const float bR = bh[j], bZ = bh[H + j], bN = bh[2 * H + j];
#pragma unroll
      for (int i = 0; i < 8; ++i) {
        const int b = (half << 3) + i;
        const float* gi = Gi + ((size_t)b * T + tt) * (3 * H);
        const float r = sigmoidf(gi[j] + aR[i] + bR);
        const float z = sigmoidf(gi[H + j] + aZ[i] + bZ);
        const float n = tanhf(gi[2 * H + j] + r * (aN[i] + bN));
        const float hv = hst[b * H + j];
        const float hn = (1.f - z) * n + z * hv;
        hst[b * H + j] = hn;
        float* o = out + (size_t)b * obstride + (size_t)tt * ostride + ocol + j;
        if (accum) *o += hn; else *o = hn;
      }
    }
    __syncthreads();
  }
}

// ---- word embedding gather: hword = s[start] + s[end] + bert ---------------
__global__ void k_word_emb(const float* __restrict__ s, const int* __restrict__ gate,
                           const float* __restrict__ bert, const int* __restrict__ idxl,
                           float* __restrict__ hw) {
  const int bw = blockIdx.x;
  const int b = bw / kW, w = bw % kW;
  const int st = gate[b * (kW + 1) + w];
  const int en = gate[b * (kW + 1) + w + 1] - 1;
  const int bi = idxl[b];
  const float* s1 = s + ((size_t)b * kL + st) * kGRUH;
  const float* s2 = s + ((size_t)b * kL + en) * kGRUH;
  const float* bt = bert + ((size_t)bi * kW + w) * kGRUH;
  float* o = hw + (size_t)bw * kGRUH;
  for (int j = threadIdx.x; j < kGRUH; j += blockDim.x) o[j] = s1[j] + s2[j] + bt[j];
}

// ---- conv bank (k = 1..8, channels-last, direct fp32: <5% of FLOPs) --------
struct BankW { const float* w[8]; const float* b[8]; };
__global__ void k_bank(const float* __restrict__ x, BankW a, float* __restrict__ y) {
  const int idx = blockIdx.x * blockDim.x + threadIdx.x;  // B*W*1024 exact
  const int co = idx & 1023, bt = idx >> 10;
  const int b = bt / kW, t = bt % kW;
  const int k = co >> 7, c = co & 127, kk = k + 1, lo = kk >> 1;
  const float* wv = a.w[k] + (size_t)c * kPREO * kk;  // w[c][ci][j]
  float acc = a.b[k][c];
  for (int j = 0; j < kk; ++j) {
    const int ts = t + j - lo;
    if (ts < 0 || ts >= kW) continue;
    const float* xr = x + ((size_t)b * kW + ts) * kPREO;
    for (int ci = 0; ci < kPREO; ++ci) acc += wv[ci * kk + j] * xr[ci];
  }
  y[(size_t)bt * (kBANKK * kBANKC) + co] = fmaxf(acc, 0.f);
}

// maxpool width 2, stride 1, pad (0,1) over time
__global__ void k_pool(const float* __restrict__ y, float* __restrict__ o) {
  const int idx = blockIdx.x * blockDim.x + threadIdx.x;  // B*W*1024 exact
  const int bt = idx >> 10, t = bt % kW;
  float v = y[idx];
  if (t + 1 < kW) v = fmaxf(v, y[idx + 1024]);
  o[idx] = v;
}

// k=3 conv, pad (1,1), optional relu, optional residual add
__global__ void k_proj3(const float* __restrict__ x, int Cin,
                        const float* __restrict__ w, const float* __restrict__ bias,
                        int Cout, const float* __restrict__ resid, int act,
                        float* __restrict__ y) {
  const int idx = blockIdx.x * blockDim.x + threadIdx.x;  // B*W*Cout exact
  const int co = idx % Cout, bt = idx / Cout;
  const int b = bt / kW, t = bt % kW;
  float acc = bias[co];
  for (int j = 0; j < 3; ++j) {
    const int ts = t + j - 1;
    if (ts < 0 || ts >= kW) continue;
    const float* xr = x + ((size_t)b * kW + ts) * Cin;
    const float* wr = w + (size_t)co * Cin * 3 + j;
    for (int ci = 0; ci < Cin; ++ci) acc += wr[ci * 3] * xr[ci];
  }
  if (act) acc = fmaxf(acc, 0.f);
  if (resid) acc += resid[(size_t)bt * Cout + co];
  y[(size_t)bt * Cout + co] = acc;
}

// highway combine: y = Hh*T + y*(1-T)
__global__ void k_hwmix(float* __restrict__ y, const float* __restrict__ Hh,
                        const float* __restrict__ Tg, int n) {
  const int i = blockIdx.x * blockDim.x + threadIdx.x;
  if (i < n) { const float t = Tg[i]; y[i] = Hh[i] * t + y[i] * (1.f - t); }
}

// ---------------------------------------------------------------------------
// Host orchestration
// ---------------------------------------------------------------------------
namespace {
struct Bump {
  char* base; size_t off;
  template <typename T> T* get(size_t n) {
    T* p = (T*)(base + off);
    off = (off + n * sizeof(T) + 255) & ~(size_t)255;
    return p;
  }
};
}

extern "C" void kernel_launch(void* const* d_in, const int* in_sizes, int n_in,
                              void* d_out, int out_size, void* d_ws, size_t ws_size,
                              hipStream_t stream) {
  (void)in_sizes; (void)n_in; (void)out_size; (void)ws_size;
  const int TPB = 256;

  // ---- inputs (jax pytree order: top-level insertion, params sorted keys) --
  const int*   x    = (const int*)d_in[0];
  const float* bert = (const float*)d_in[1];
  const int*   gate = (const int*)d_in[2];
  const int*   idxl = (const int*)d_in[3];
  const float *bank_b[8], *bank_w[8];
  for (int k = 0; k < 8; ++k) { bank_b[k] = (const float*)d_in[4 + 2 * k];
                                bank_w[k] = (const float*)d_in[5 + 2 * k]; }
  // cbhg_gru: b {Wh,Wi,bh,bi}, f {...}
  const float* cgb_Wh = (const float*)d_in[20]; const float* cgb_Wi = (const float*)d_in[21];
  const float* cgb_bh = (const float*)d_in[22]; const float* cgb_bi = (const float*)d_in[23];
  const float* cgf_Wh = (const float*)d_in[24]; const float* cgf_Wi = (const float*)d_in[25];
  const float* cgf_bh = (const float*)d_in[26]; const float* cgf_bi = (const float*)d_in[27];
  const float* embtab = (const float*)d_in[28];
  const float* egb_Wh = (const float*)d_in[29]; const float* egb_Wi = (const float*)d_in[30];
  const float* egb_bh = (const float*)d_in[31]; const float* egb_bi = (const float*)d_in[32];
  const float* egf_Wh = (const float*)d_in[33]; const float* egf_Wi = (const float*)d_in[34];
  const float* egf_bh = (const float*)d_in[35]; const float* egf_bi = (const float*)d_in[36];
  const float *hw_Wh[4], *hw_Wt[4], *hw_bh[4], *hw_bt[4];
  for (int i = 0; i < 4; ++i) {
    hw_Wh[i] = (const float*)d_in[37 + 4 * i]; hw_Wt[i] = (const float*)d_in[38 + 4 * i];
    hw_bh[i] = (const float*)d_in[39 + 4 * i]; hw_bt[i] = (const float*)d_in[40 + 4 * i];
  }
  const float* pgb_Wh = (const float*)d_in[53]; const float* pgb_Wi = (const float*)d_in[54];
  const float* pgb_bh = (const float*)d_in[55]; const float* pgb_bi = (const float*)d_in[56];
  const float* pgf_Wh = (const float*)d_in[57]; const float* pgf_Wi = (const float*)d_in[58];
  const float* pgf_bh = (const float*)d_in[59]; const float* pgf_bi = (const float*)d_in[60];
  const float* pn_W1 = (const float*)d_in[61]; const float* pn_W2 = (const float*)d_in[62];
  const float* pn_b1 = (const float*)d_in[63]; const float* pn_b2 = (const float*)d_in[64];
  const float* p1_b = (const float*)d_in[65]; const float* p1_w = (const float*)d_in[66];
  const float* p2_b = (const float*)d_in[67]; const float* p2_w = (const float*)d_in[68];

  // ---- workspace layout ----------------------------------------------------
  Bump ws{(char*)d_ws, 0};
  float*          embF  = ws.get<float>((size_t)kB * kL * kEMB);
  unsigned short* embBF = ws.get<unsigned short>((size_t)kB * kL * kEMB);
  float*          sm    = ws.get<float>((size_t)kB * kL);
  float*          em    = ws.get<float>((size_t)kB * kL);
  float*          sbuf  = ws.get<float>((size_t)kB * kL * kGRUH);
  float*          Gi    = ws.get<float>((size_t)kB * kL * 3 * kGRUH);  // reused for all GRUs
  float*          hword = ws.get<float>((size_t)kB * kW * kGRUH);
  unsigned short* hwBF  = ws.get<unsigned short>((size_t)kB * kW * kGRUH);
  float*          h1    = ws.get<float>((size_t)kB * kW * kPREH);
  unsigned short* h1BF  = ws.get<unsigned short>((size_t)kB * kW * kPREH);
  float*          hpre  = ws.get<float>((size_t)kB * kW * kPREO);
  float*          bank  = ws.get<float>((size_t)kB * kW * kBANKK * kBANKC);
  float*          pool  = ws.get<float>((size_t)kB * kW * kBANKK * kBANKC);
  float*          y1    = ws.get<float>((size_t)kB * kW * kPREO);
  float*          ybuf  = ws.get<float>((size_t)kB * kW * kPREO);
  unsigned short* ybf   = ws.get<unsigned short>((size_t)kB * kW * kPREO);
  float*          tmpH  = ws.get<float>((size_t)kB * kW * kPREO);
  float*          tmpT  = ws.get<float>((size_t)kB * kW * kPREO);
  // bf16 weights
  unsigned short* pgfWi = ws.get<unsigned short>(2304 * 256);
  unsigned short* pgfWh = ws.get<unsigned short>(2304 * 768);
  unsigned short* pgbWi = ws.get<unsigned short>(2304 * 256);
  unsigned short* pgbWh = ws.get<unsigned short>(2304 * 768);
  unsigned short* egfWi = ws.get<unsigned short>(768 * 256);
  unsigned short* egfWh = ws.get<unsigned short>(768 * 256);
  unsigned short* egbWi = ws.get<unsigned short>(768 * 256);
  unsigned short* egbWh = ws.get<unsigned short>(768 * 256);
  unsigned short* cgfWi = ws.get<unsigned short>(384 * 256);
  unsigned short* cgfWh = ws.get<unsigned short>(384 * 128);
  unsigned short* cgbWi = ws.get<unsigned short>(384 * 256);
  unsigned short* cgbWh = ws.get<unsigned short>(384 * 128);
  unsigned short* W1bf  = ws.get<unsigned short>(1024 * 768);
  unsigned short* W2bf  = ws.get<unsigned short>(256 * 1024);
  unsigned short* hwWhB[4]; unsigned short* hwWtB[4];
  for (int i = 0; i < 4; ++i) { hwWhB[i] = ws.get<unsigned short>(256 * 256);
                                hwWtB[i] = ws.get<unsigned short>(256 * 256); }

  auto cvt = [&](const float* s, unsigned short* d, int n) {
    k_cvt_bf16<<<(n + TPB - 1) / TPB, TPB, 0, stream>>>(s, d, n);
  };
  auto gemm = [&](const unsigned short* A, int lda, const unsigned short* Wt, int ldw,
                  const float* bias, float* C, int ldc, int M, int N, int K, int act) {
    long tiles = (long)(M / 16) * (N / 16);
    int blocks = (int)((tiles + 7) / 8);
    k_gemm_bf16<<<blocks, 256, 0, stream>>>(A, lda, Wt, ldw, bias, C, ldc, M, N, K, act);
  };

  // ---- 0: weight conversions ----------------------------------------------
  cvt(pgf_Wi, pgfWi, 2304 * 256); cvt(pgf_Wh, pgfWh, 2304 * 768);
  cvt(pgb_Wi, pgbWi, 2304 * 256); cvt(pgb_Wh, pgbWh, 2304 * 768);
  cvt(egf_Wi, egfWi, 768 * 256);  cvt(egf_Wh, egfWh, 768 * 256);
  cvt(egb_Wi, egbWi, 768 * 256);  cvt(egb_Wh, egbWh, 768 * 256);
  cvt(cgf_Wi, cgfWi, 384 * 256);  cvt(cgf_Wh, cgfWh, 384 * 128);
  cvt(cgb_Wi, cgbWi, 384 * 256);  cvt(cgb_Wh, cgbWh, 384 * 128);
  cvt(pn_W1, W1bf, 1024 * 768);   cvt(pn_W2, W2bf, 256 * 1024);
  for (int i = 0; i < 4; ++i) { cvt(hw_Wh[i], hwWhB[i], 256 * 256);
                                cvt(hw_Wt[i], hwWtB[i], 256 * 256); }

  // ---- 1: embedding + reset masks -----------------------------------------
  k_embed<<<(kB * kL * kEMB) / TPB, TPB, 0, stream>>>(x, embtab, embF, embBF);
  k_zero<<<(2 * kB * kL + TPB - 1) / TPB, TPB, 0, stream>>>(sm, 2 * kB * kL);
  k_masks<<<(kB * kW + TPB - 1) / TPB, TPB, 0, stream>>>(gate, sm, em);

  // ---- 2: pre_gru (H=768, T=512), s = f + b --------------------------------
  gemm(embBF, 256, pgfWi, 256, pgf_bi, Gi, 2304, kB * kL, 2304, 256, 0);
  k_gru_scan<768><<<1, 1024, 0, stream>>>(Gi, pgfWh, pgf_bh, sm, kL, 0,
                                          sbuf, (long)kL * kGRUH, kGRUH, 0, 0);
  gemm(embBF, 256, pgbWi, 256, pgb_bi, Gi, 2304, kB * kL, 2304, 256, 0);
  k_gru_scan<768><<<1, 1024, 0, stream>>>(Gi, pgbWh, pgb_bh, em, kL, 1,
                                          sbuf, (long)kL * kGRUH, kGRUH, 0, 1);

  // ---- 3: word emb gather + prenet ----------------------------------------
  k_word_emb<<<kB * kW, 256, 0, stream>>>(sbuf, gate, bert, idxl, hword);
  cvt(hword, hwBF, kB * kW * kGRUH);
  gemm(hwBF, 768, W1bf, 768, pn_b1, h1, 1024, kB * kW, 1024, 768, 1);
  cvt(h1, h1BF, kB * kW * kPREH);
  gemm(h1BF, 1024, W2bf, 1024, pn_b2, hpre, 256, kB * kW, 256, 1024, 1);

  // ---- 4: conv bank + pool + projections -----------------------------------
  BankW bw;
  for (int k = 0; k < 8; ++k) { bw.w[k] = bank_w[k]; bw.b[k] = bank_b[k]; }
  k_bank<<<(kB * kW * 1024) / TPB, TPB, 0, stream>>>(hpre, bw, bank);
  k_pool<<<(kB * kW * 1024) / TPB, TPB, 0, stream>>>(bank, pool);
  k_proj3<<<(kB * kW * 256) / TPB, TPB, 0, stream>>>(pool, 1024, p1_w, p1_b, 256,
                                                     nullptr, 1, y1);
  k_proj3<<<(kB * kW * 256) / TPB, TPB, 0, stream>>>(y1, 256, p2_w, p2_b, 256,
                                                     hpre, 0, ybuf);

  // ---- 5: highway x4 -------------------------------------------------------
  const int nY = kB * kW * kPREO;
  for (int i = 0; i < 4; ++i) {
    cvt(ybuf, ybf, nY);
    gemm(ybf, 256, hwWhB[i], 256, hw_bh[i], tmpH, 256, kB * kW, 256, 256, 1);
    gemm(ybf, 256, hwWtB[i], 256, hw_bt[i], tmpT, 256, kB * kW, 256, 256, 2);
    k_hwmix<<<(nY + TPB - 1) / TPB, TPB, 0, stream>>>(ybuf, tmpH, tmpT, nY);
  }

  // ---- 6: cbhg_gru (H=128, T=64) -> out word = concat(f,b) -----------------
  float* out_word  = (float*)d_out;                          // (B, W, 256)
  float* out_alpha = (float*)d_out + (size_t)kB * kW * 256;  // (B, L, 256)
  cvt(ybuf, ybf, nY);
  gemm(ybf, 256, cgfWi, 256, cgf_bi, Gi, 384, kB * kW, 384, 256, 0);
  k_gru_scan<128><<<1, 1024, 0, stream>>>(Gi, cgfWh, cgf_bh, nullptr, kW, 0,
                                          out_word, (long)kW * 256, 256, 0, 0);
  gemm(ybf, 256, cgbWi, 256, cgb_bi, Gi, 384, kB * kW, 384, 256, 0);
  k_gru_scan<128><<<1, 1024, 0, stream>>>(Gi, cgbWh, cgb_bh, nullptr, kW, 1,
                                          out_word, (long)kW * 256, 256, 128, 0);

  // ---- 7: emb_gru (H=256, T=512) -> out alpha = f + b ----------------------
  gemm(embBF, 256, egfWi, 256, egf_bi, Gi, 768, kB * kL, 768, 256, 0);
  k_gru_scan<256><<<1, 1024, 0, stream>>>(Gi, egfWh, egf_bh, nullptr, kL, 0,
                                          out_alpha, (long)kL * 256, 256, 0, 0);
  gemm(embBF, 256, egbWi, 256, egb_bi, Gi, 768, kB * kL, 768, 256, 0);
  k_gru_scan<256><<<1, 1024, 0, stream>>>(Gi, egbWh, egb_bh, nullptr, kL, 1,
                                          out_alpha, (long)kL * 256, 256, 0, 1);
}
